// GATv2Convolution_75333726372327
// MI455X (gfx1250) — compile-verified, matrix operand order checked by reading, not run on hardware
//
#include <hip/hip_runtime.h>
#include <hip/hip_bf16.h>
#include <math.h>

// ---------------------------------------------------------------------------
// GATv2 convolution, restructured:
//   P = h @ W[0:64,:]   (node-level, WMMA bf16)   -> gathered by src
//   Q = h @ W[64:128,:] (node-level, WMMA bf16)   -> gathered by tgt
//   logit[e,h] = sum_d lrelu(P[src,h*8+d]+Q[tgt,h*8+d]) * a[h*8+d]
//   segment softmax over tgt (atomic max / atomic sum), then
//   out[tgt] += h[src] * att   (per-head scatter)
// ---------------------------------------------------------------------------

typedef __attribute__((ext_vector_type(16))) __bf16 v16bf;
typedef __attribute__((ext_vector_type(8)))  float  v8f;

#define IN_CH   64
#define HEADS   8
#define OUT_CH  8
#define HD      (HEADS * OUT_CH)   // 64
#define LEAKY   0.3f
#define HT_PITCH 68                // padded LDS row stride (banks spread)

// ---------------------------------------------------------------------------
// Kernel A: dual projection GEMM via WMMA bf16 (f32 accumulate).
// Grid: nNodes/16 blocks of 256 threads (8 waves).
// Wave w: mat = w>>2 (0=P uses W rows 0..63, 1=Q uses W rows 64..127),
//         tn  = w&3  (16-wide output column tile).
// A tile (16x64 of h) and the whole W (128x64) are staged in LDS with
// coalesced float4 loads; all fragment builds then read LDS.
// NOTE: no divergence — WMMA requires EXEC all-ones; nNodes (50000) is an
//       exact multiple of 16 so the grid tiles exactly.
// ---------------------------------------------------------------------------
__global__ __launch_bounds__(256) void gat_project_wmma(
    const float* __restrict__ h, const float* __restrict__ W,
    float* __restrict__ P, float* __restrict__ Q, int nNodes) {
  __shared__ float hT[16 * HT_PITCH];   // 16 rows x 64 ch, padded pitch
  __shared__ float Wl[2 * IN_CH * HD];  // full W, 128 x 64

  const int tid = threadIdx.x;

  // Stage h tile: 1024 floats, one float4 per thread (coalesced).
  {
    const float4* s4 = (const float4*)(h + (size_t)blockIdx.x * 16 * IN_CH);
    const int r = tid >> 4;            // row 0..15
    const int c = (tid & 15) * 4;      // col 0,4,...,60
    *(float4*)(hT + r * HT_PITCH + c) = s4[tid];
  }
  // Stage W: 8192 floats = 2048 float4, 8 per thread (coalesced).
  {
    const float4* s4 = (const float4*)W;
    float4* d4 = (float4*)Wl;
    #pragma unroll
    for (int i = 0; i < 8; ++i)
      d4[tid + 256 * i] = s4[tid + 256 * i];
  }
  __syncthreads();

  const int wave  = tid >> 5;
  const int lane  = tid & 31;
  const int mat   = wave >> 2;        // 0 -> P, 1 -> Q
  const int tn    = wave & 3;         // output col tile (16 cols)
  const int l16   = lane & 15;
  const int khalf = lane >> 4;        // 0: K 0..7/16..23, 1: K 8..15/24..31
  const int col   = tn * 16 + l16;    // B-matrix col for this lane
  const int wbase = mat * IN_CH;      // W row offset (0 or 64)

  const float* __restrict__ hrow = hT + l16 * HT_PITCH;

  v8f acc = {};
  #pragma unroll
  for (int kb = 0; kb < IN_CH; kb += 32) {      // two K-blocks of 32
    const int k0 = kb + khalf * 8;
    v16bf A, B;
    #pragma unroll
    for (int j = 0; j < 8; ++j) {
      // A (16x32 bf16): lane half selects K sub-range per ISA layout
      A[j]     = (__bf16)hrow[k0 + j];
      A[j + 8] = (__bf16)hrow[k0 + 16 + j];
      // B (32x16 bf16): lane holds one output column, same K striping
      B[j]     = (__bf16)Wl[(wbase + k0 + j)      * HD + col];
      B[j + 8] = (__bf16)Wl[(wbase + k0 + 16 + j) * HD + col];
    }
    acc = __builtin_amdgcn_wmma_f32_16x16x32_bf16(
        /*neg_a=*/false, A, /*neg_b=*/false, B,
        /*c_mod=*/(short)0, acc, /*reuse_a=*/false, /*reuse_b=*/false);
  }

  // C/D layout: VGPR v, lanes 0-15 -> row v, lanes 16-31 -> row v+8
  float* __restrict__ dst = mat ? Q : P;
  const int rbase = blockIdx.x * 16 + khalf * 8;
  #pragma unroll
  for (int v = 0; v < 8; ++v)
    dst[(size_t)(rbase + v) * HD + col] = acc[v];
}

// ---------------------------------------------------------------------------
// Kernel B: zero/init accumulators. out[N*64]=0, m[N*8]=-inf, s[N*8]=0.
// ---------------------------------------------------------------------------
__global__ __launch_bounds__(256) void gat_init(
    float* __restrict__ out, float* __restrict__ m, float* __restrict__ s,
    int n64, int n8) {
  int i = blockIdx.x * blockDim.x + threadIdx.x;
  if (i < n64) out[i] = 0.0f;
  if (i < n8) { m[i] = -__builtin_inff(); s[i] = 0.0f; }
}

// ---------------------------------------------------------------------------
// Kernel C: per (edge, head) logit + atomic segment-max over tgt.
// Float max via int/uint punning (bit-deterministic, unlike float add).
// ---------------------------------------------------------------------------
__global__ __launch_bounds__(256) void gat_logits(
    const float* __restrict__ P, const float* __restrict__ Q,
    const float* __restrict__ a,
    const int* __restrict__ src, const int* __restrict__ tgt,
    float* __restrict__ logits, float* __restrict__ m, int nEdges) {
  int t = blockIdx.x * blockDim.x + threadIdx.x;
  if (t >= nEdges * HEADS) return;
  const int e  = t >> 3;
  const int hd = t & 7;
  const int sn = src[e];
  const int tn = tgt[e];
  const float* __restrict__ p  = P + (size_t)sn * HD + hd * OUT_CH;
  const float* __restrict__ q  = Q + (size_t)tn * HD + hd * OUT_CH;
  const float* __restrict__ av = a + hd * OUT_CH;
  float lg = 0.0f;
  #pragma unroll
  for (int d = 0; d < OUT_CH; ++d) {
    float x = p[d] + q[d];
    x = x > 0.0f ? x : LEAKY * x;    // leaky_relu
    lg += x * av[d];
  }
  logits[t] = lg;
  float* addr = m + (size_t)tn * HEADS + hd;
  if (lg >= 0.0f) atomicMax((int*)addr, __float_as_int(lg));
  else            atomicMin((unsigned int*)addr, (unsigned int)__float_as_int(lg));
}

// ---------------------------------------------------------------------------
// Kernel D: ex = exp(logit - m[tgt]); atomic segment-sum into s.
// ---------------------------------------------------------------------------
__global__ __launch_bounds__(256) void gat_expsum(
    const int* __restrict__ tgt, const float* __restrict__ m,
    float* __restrict__ logits, float* __restrict__ s, int nEdges) {
  int t = blockIdx.x * blockDim.x + threadIdx.x;
  if (t >= nEdges * HEADS) return;
  const int e  = t >> 3;
  const int hd = t & 7;
  const int tn = tgt[e];
  float ex = __expf(logits[t] - m[(size_t)tn * HEADS + hd]);
  logits[t] = ex;                           // reuse buffer: logits -> ex
  atomicAdd(s + (size_t)tn * HEADS + hd, ex);
}

// ---------------------------------------------------------------------------
// Kernel E: att = ex / s[tgt]; out[tgt] += h[src] * att per head (scatter).
// ---------------------------------------------------------------------------
__global__ __launch_bounds__(256) void gat_scatter(
    const float* __restrict__ h,
    const int* __restrict__ src, const int* __restrict__ tgt,
    const float* __restrict__ ex, const float* __restrict__ s,
    float* __restrict__ out, int nEdges) {
  int t = blockIdx.x * blockDim.x + threadIdx.x;
  if (t >= nEdges * HEADS) return;
  const int e  = t >> 3;
  const int hd = t & 7;
  const int sn = src[e];
  const int tn = tgt[e];
  const float att = ex[t] / s[(size_t)tn * HEADS + hd];
  const float* __restrict__ hi = h + (size_t)sn * IN_CH + hd * OUT_CH;
  float* __restrict__ o = out + (size_t)tn * HD + hd * OUT_CH;
  #pragma unroll
  for (int d = 0; d < OUT_CH; ++d)
    atomicAdd(o + d, hi[d] * att);
}

// ---------------------------------------------------------------------------
// Launch
// ---------------------------------------------------------------------------
extern "C" void kernel_launch(void* const* d_in, const int* in_sizes, int n_in,
                              void* d_out, int out_size, void* d_ws, size_t ws_size,
                              hipStream_t stream) {
  const float* h   = (const float*)d_in[0];   // [N, 64]
  const float* W   = (const float*)d_in[1];   // [128, 64]
  const float* a   = (const float*)d_in[2];   // [1, 64]
  const int*   src = (const int*)d_in[3];     // [E]
  const int*   tgt = (const int*)d_in[4];     // [E]
  float*       out = (float*)d_out;           // [N, 64]

  const int nNodes = in_sizes[0] / IN_CH;     // 50000
  const int nEdges = in_sizes[3];             // 800000

  // Workspace layout (floats):
  //   P[N*64] | Q[N*64] | logits/ex[E*8] | m[N*8] | s[N*8]   (~54 MB)
  float* P      = (float*)d_ws;
  float* Q      = P + (size_t)nNodes * HD;
  float* logits = Q + (size_t)nNodes * HD;
  float* m      = logits + (size_t)nEdges * HEADS;
  float* s      = m + (size_t)nNodes * HEADS;

  // A: WMMA projection (nNodes multiple of 16 -> exact tiling, no divergence)
  const int rowBlocks = nNodes / 16;
  gat_project_wmma<<<rowBlocks, 256, 0, stream>>>(h, W, P, Q, nNodes);

  // B: init accumulators
  const int n64 = nNodes * HD, n8 = nNodes * HEADS;
  gat_init<<<(n64 + 255) / 256, 256, 0, stream>>>(out, m, s, n64, n8);

  // C/D/E: edge phase, one thread per (edge, head)
  const int eh = nEdges * HEADS;
  const int ehBlocks = (eh + 255) / 256;
  gat_logits <<<ehBlocks, 256, 0, stream>>>(P, Q, a, src, tgt, logits, m, nEdges);
  gat_expsum <<<ehBlocks, 256, 0, stream>>>(tgt, m, logits, s, nEdges);
  gat_scatter<<<ehBlocks, 256, 0, stream>>>(h, src, tgt, logits, s, out, nEdges);
}
